// Attention_Module_46411416600754
// MI455X (gfx1250) — matrix operand order repaired and checked by reading, not compiled, not run on hardware
//
#include <hip/hip_runtime.h>
#include <hip/hip_bf16.h>
#include <math.h>

typedef __attribute__((ext_vector_type(16))) _Float16 v16h;
typedef __attribute__((ext_vector_type(8)))  float    v8f;

#define B_  32
#define L_  2
#define S_  4096
#define H_  1024
#define SC  32              // rows of S per chunk (32*1024*4 = 128KB LDS tile)
#define NC  (S_ / SC)       // 128 chunks
#define NT  (H_ / 16)       // 64 n-tiles in the WMMA kernel

// workspace layout (float offsets)
#define WS_U      0                      // [H_]
#define WS_CPART  (WS_U + H_)            // [B_ * NT]   per-(batch, n-tile) partials of c[b]
#define WS_M      (WS_CPART + B_ * NT)   // [B_ * NC]   chunk max
#define WS_Z      (WS_M + B_ * NC)       // [B_ * NC]   chunk sum
#define WS_CTX    (WS_Z + B_ * NC)       // [B_ * NC * H_] chunk context partials

// ---------------------------------------------------------------------------
// Kernel A: u[h] = sum_o V[o] * U[o,h]
// ---------------------------------------------------------------------------
__global__ void k_uvec(const float* __restrict__ U, const float* __restrict__ V,
                       float* __restrict__ ws) {
  int h = blockIdx.x * blockDim.x + threadIdx.x;
  if (h >= H_) return;
  float acc = 0.f;
  for (int o = 0; o < H_; ++o) acc = fmaf(V[o], U[(size_t)o * H_ + h], acc);
  ws[WS_U + h] = acc;
}

// ---------------------------------------------------------------------------
// Kernel B: c-partials via WMMA.  T = q @ W^T (M=32,N=1024,K=1024), then
// cpart[b, nt] = sum over the 16 n of tanh(T[b,n]) * V[n].
// One wave per 16x16 output tile; K-loop of 32 x (16x16x32 f16 WMMA).
// ---------------------------------------------------------------------------
__global__ __launch_bounds__(32)
void k_wq_wmma(const float* __restrict__ query, const float* __restrict__ W,
               const float* __restrict__ V, float* __restrict__ ws) {
  const int lane  = threadIdx.x;      // 0..31
  const int tile  = blockIdx.x;       // 0..127
  const int mt    = tile >> 6;        // 0..1   (batch tile)
  const int nt    = tile & 63;        // 0..63  (output-feature tile)
  const int m0    = mt * 16, n0 = nt * 16;
  const int row16 = lane & 15;
  const int hi    = lane >> 4;        // half-wave id

  // q row for this lane's M, last L index: query[b, L-1, :]
  const float* qrow = query + (size_t)(m0 + row16) * (L_ * H_) + (size_t)(L_ - 1) * H_;
  const float* wrow = W + (size_t)(n0 + row16) * H_;

  v8f c = {};
  for (int k0 = 0; k0 < H_; k0 += 32) {
    // A 16x32 f16: lane<16 holds K runs [0..7] and [16..23]; lane>=16: +8
    const int ka = k0 + hi * 8;
    const float4* q4 = reinterpret_cast<const float4*>(qrow + ka);
    float4 a0 = q4[0], a1 = q4[1];     // ka .. ka+7
    float4 a2 = q4[4], a3 = q4[5];     // ka+16 .. ka+23
    v16h a;
    a[0]=(_Float16)a0.x; a[1]=(_Float16)a0.y; a[2]=(_Float16)a0.z; a[3]=(_Float16)a0.w;
    a[4]=(_Float16)a1.x; a[5]=(_Float16)a1.y; a[6]=(_Float16)a1.z; a[7]=(_Float16)a1.w;
    a[8]=(_Float16)a2.x; a[9]=(_Float16)a2.y; a[10]=(_Float16)a2.z; a[11]=(_Float16)a2.w;
    a[12]=(_Float16)a3.x; a[13]=(_Float16)a3.y; a[14]=(_Float16)a3.z; a[15]=(_Float16)a3.w;

    // B 32x16 f16: lane<16 holds K runs [0..7] and [16..23] of column n; lane>=16: +8
    const int kb = k0 + hi * 8;
    const float4* w4 = reinterpret_cast<const float4*>(wrow + kb);
    float4 b0 = w4[0], b1 = w4[1], b2 = w4[4], b3 = w4[5];
    v16h b;
    b[0]=(_Float16)b0.x; b[1]=(_Float16)b0.y; b[2]=(_Float16)b0.z; b[3]=(_Float16)b0.w;
    b[4]=(_Float16)b1.x; b[5]=(_Float16)b1.y; b[6]=(_Float16)b1.z; b[7]=(_Float16)b1.w;
    b[8]=(_Float16)b2.x; b[9]=(_Float16)b2.y; b[10]=(_Float16)b2.z; b[11]=(_Float16)b2.w;
    b[12]=(_Float16)b3.x; b[13]=(_Float16)b3.y; b[14]=(_Float16)b3.z; b[15]=(_Float16)b3.w;

    c = __builtin_amdgcn_wmma_f32_16x16x32_f16(false, a, false, b, (short)0, c,
                                               false, false);
  }

  const float vn = V[n0 + row16];
  #pragma unroll
  for (int j = 0; j < 8; ++j) {
    // C layout: VGPR j -> (M = j + 8*hi, N = lane&15)
    float v = tanhf(c[j]) * vn;
    v += __shfl_xor(v, 1);
    v += __shfl_xor(v, 2);
    v += __shfl_xor(v, 4);
    v += __shfl_xor(v, 8);          // sum over 16 lanes of this half-wave
    if (row16 == 0) {
      int m = m0 + j + hi * 8;      // batch index
      ws[WS_CPART + m * NT + nt] = v;
    }
  }
}

// ---------------------------------------------------------------------------
// Kernel C: per (batch, chunk): async-stage 32x1024 keys tile into LDS,
// scores = keys.u + c[b]; chunk-local online softmax; context partial.
// ---------------------------------------------------------------------------
__global__ __launch_bounds__(256, 1)
void k_main(const float* __restrict__ keys, float* __restrict__ ws,
            float* __restrict__ out) {
  __shared__ float lds_keys[SC * H_];   // 128 KB
  __shared__ float lds_u[H_];           // 4 KB
  __shared__ float lds_score[SC];
  __shared__ float lds_e[SC];
  __shared__ float lds_cb;

  const int tid = threadIdx.x;
  const int cid = blockIdx.x;           // chunk 0..NC-1
  const int b   = blockIdx.y;           // batch
  const int s0  = cid * SC;

  // stage u into LDS (256 x float4 = 1024 floats)
  reinterpret_cast<float4*>(lds_u)[tid] =
      reinterpret_cast<const float4*>(ws + WS_U)[tid];

  if (tid == 0) {
    float cb = 0.f;
    for (int t = 0; t < NT; ++t) cb += ws[WS_CPART + b * NT + t];
    lds_cb = cb;
  }

  // async copy keys[b, s0:s0+SC, :] into LDS (GVS: SGPR base + VGPR offset)
  const float* gbase = keys + ((size_t)b * S_ + (size_t)s0) * H_;
  unsigned ldsBase = (unsigned)(unsigned long long)(&lds_keys[0]);
  #pragma unroll 8
  for (int i = 0; i < (SC * H_ / 4) / 256; ++i) {     // 32 x b128 per thread
    unsigned fid   = (unsigned)(i * 256 + tid);
    unsigned goff  = fid * 16u;
    unsigned laddr = ldsBase + fid * 16u;
    asm volatile("global_load_async_to_lds_b128 %0, %1, %2"
                 :: "v"(laddr), "v"(goff), "s"(gbase) : "memory");
  }
#if __has_builtin(__builtin_amdgcn_s_wait_asynccnt)
  __builtin_amdgcn_s_wait_asynccnt(0);
#else
  asm volatile("s_wait_asynccnt 0x0" ::: "memory");
#endif
  __syncthreads();

  // phase 1: dot products (one row per wave-iteration, shuffle reduce)
  const float4* lk4 = reinterpret_cast<const float4*>(lds_keys);
  const float4* lu4 = reinterpret_cast<const float4*>(lds_u);
  const int lane = tid & 31;
  const int wave = tid >> 5;                 // 8 waves, SC/8 = 4 rows each
  for (int r = 0; r < SC / 8; ++r) {
    int s = wave * (SC / 8) + r;
    float acc = 0.f;
    #pragma unroll
    for (int kk = 0; kk < 8; ++kk) {
      float4 kv = lk4[s * 256 + kk * 32 + lane];
      float4 uv = lu4[kk * 32 + lane];
      acc = fmaf(kv.x, uv.x, acc);
      acc = fmaf(kv.y, uv.y, acc);
      acc = fmaf(kv.z, uv.z, acc);
      acc = fmaf(kv.w, uv.w, acc);
    }
    acc += __shfl_xor(acc, 16);
    acc += __shfl_xor(acc, 8);
    acc += __shfl_xor(acc, 4);
    acc += __shfl_xor(acc, 2);
    acc += __shfl_xor(acc, 1);
    if (lane == 0) lds_score[s] = acc + lds_cb;
  }
  __syncthreads();

  // phase 2: chunk max / exp / sum
  float m = -1e30f;
  for (int s = 0; s < SC; ++s) m = fmaxf(m, lds_score[s]);
  if (tid < SC) {
    float e = expf(lds_score[tid] - m);
    lds_e[tid] = e;
    out[(size_t)B_ * H_ + (size_t)b * S_ + s0 + tid] = e;   // chunk-normalized
  }
  if (tid == 0) {
    float z = 0.f;
    for (int s = 0; s < SC; ++s) z += expf(lds_score[s] - m);
    ws[WS_M + b * NC + cid] = m;
    ws[WS_Z + b * NC + cid] = z;
  }
  __syncthreads();

  // phase 3: context partial, each thread owns 4 h-columns
  float4 acc4 = make_float4(0.f, 0.f, 0.f, 0.f);
  for (int s = 0; s < SC; ++s) {
    float  e  = lds_e[s];
    float4 kv = lk4[s * 256 + tid];
    acc4.x = fmaf(e, kv.x, acc4.x);
    acc4.y = fmaf(e, kv.y, acc4.y);
    acc4.z = fmaf(e, kv.z, acc4.z);
    acc4.w = fmaf(e, kv.w, acc4.w);
  }
  reinterpret_cast<float4*>(ws + WS_CTX)[((size_t)b * NC + cid) * 256 + tid] = acc4;
}

// ---------------------------------------------------------------------------
// Kernel D: combine chunk partials per batch; finalize context + weights.
// ---------------------------------------------------------------------------
__global__ __launch_bounds__(256)
void k_combine(float* __restrict__ ws, float* __restrict__ out) {
  __shared__ float sf[NC];
  const int tid = threadIdx.x;
  const int b   = blockIdx.x;

  float M = -1e30f;
  for (int c = 0; c < NC; ++c) M = fmaxf(M, ws[WS_M + b * NC + c]);
  float Z = 0.f;
  for (int c = 0; c < NC; ++c)
    Z += ws[WS_Z + b * NC + c] * expf(ws[WS_M + b * NC + c] - M);
  if (tid < NC) sf[tid] = expf(ws[WS_M + b * NC + tid] - M);
  __syncthreads();
  const float invZ = 1.f / Z;

  // context
  const float4* ctx4 = reinterpret_cast<const float4*>(ws + WS_CTX);
  float4 acc = make_float4(0.f, 0.f, 0.f, 0.f);
  for (int c = 0; c < NC; ++c) {
    float  f = sf[c];
    float4 v = ctx4[((size_t)b * NC + c) * 256 + tid];
    acc.x = fmaf(f, v.x, acc.x);
    acc.y = fmaf(f, v.y, acc.y);
    acc.z = fmaf(f, v.z, acc.z);
    acc.w = fmaf(f, v.w, acc.w);
  }
  acc.x *= invZ; acc.y *= invZ; acc.z *= invZ; acc.w *= invZ;
  reinterpret_cast<float4*>(out)[(size_t)b * 256 + tid] = acc;

  // weights: rescale chunk-normalized exp values in place
  float* wout = out + (size_t)B_ * H_ + (size_t)b * S_;
  for (int r = 0; r < S_ / 256; ++r) {
    int s = r * 256 + tid;
    wout[s] = wout[s] * sf[s / SC] * invZ;
  }
}

// ---------------------------------------------------------------------------
extern "C" void kernel_launch(void* const* d_in, const int* in_sizes, int n_in,
                              void* d_out, int out_size, void* d_ws, size_t ws_size,
                              hipStream_t stream) {
  (void)in_sizes; (void)n_in; (void)out_size; (void)ws_size;
  const float* query = (const float*)d_in[0];
  const float* keys  = (const float*)d_in[1];
  const float* W     = (const float*)d_in[2];
  const float* U     = (const float*)d_in[3];
  const float* V     = (const float*)d_in[4];
  float* out = (float*)d_out;
  float* ws  = (float*)d_ws;

  k_uvec   <<<dim3((H_ + 255) / 256), dim3(256), 0, stream>>>(U, V, ws);
  k_wq_wmma<<<dim3(2 * NT),           dim3(32),  0, stream>>>(query, W, V, ws);
  k_main   <<<dim3(NC, B_),           dim3(256), 0, stream>>>(keys, ws, out);
  k_combine<<<dim3(B_),               dim3(256), 0, stream>>>(ws, out);
}
